// KLinear_37769942401395
// MI455X (gfx1250) — compile-verified, matrix operand order checked by reading, not run on hardware
//
#include <hip/hip_runtime.h>

// ---------------------------------------------------------------------------
// RBF layer fused kernel for MI455X (gfx1250, wave32, WMMA, async-to-LDS,
// software-pipelined double buffering).
//   out = exp(-(|x|^2 + |c|^2 - 2 x c^T)) @ values
// Pre-passes (one-time, ~150MB ~ 6us @ 23.3TB/s): row norms; f16 conversion
// of x / centers; values converted f16 + transposed [n][k] for B-fragments.
// Main kernel: per 64-row x 512-col output block, loop centers in tiles of 64:
//   GEMM1 (x @ c^T) -> exp -> GEMM2 (P @ values), all f16 WMMA, f32 accum.
// All tile staging is GLOBAL_LOAD_ASYNC_TO_LDS_B128 (ASYNCcnt), issued one
// batch ahead so DMA overlaps WMMA/exp at every step.
// ---------------------------------------------------------------------------

typedef _Float16 h16;
typedef __attribute__((ext_vector_type(16))) _Float16 v16h;
typedef __attribute__((ext_vector_type(8)))  float    v8f;
typedef __attribute__((ext_vector_type(2)))  _Float16 h2v;
typedef __attribute__((ext_vector_type(4)))  float    vf4;
typedef __attribute__((ext_vector_type(4)))  unsigned int vu4;
typedef __attribute__((ext_vector_type(4)))  int      i4v;

#define DIM_IN   1024
#define DIM_OUT  1024
#define NB       8192      // batch rows
#define NK       8192      // centers
#define WROW     2048      // w row stride (floats)

#define BM       64        // rows of x per workgroup
#define BN       512       // output cols per workgroup
#define BK       64        // centers per k-iteration
#define DSTEP    64        // dim_in chunk per GEMM1 inner step
#define NSTEPS   (DIM_IN / DSTEP)   // 16
#define NCHUNK   256       // values cols staged in LDS at a time
#define LSTR     72        // LDS stride (halfs), 144B = 16B-aligned, bank-padded
#define VSTR     72        // LDS stride (halfs) for vB [NCHUNK][VSTR]

#if __has_builtin(__builtin_amdgcn_global_load_async_to_lds_b128)
#define ASYNC_LDS 1
#else
#define ASYNC_LDS 0
#endif

// per-lane 16-byte global->LDS copy (ASYNCcnt-tracked DMA, no VGPR data)
__device__ __forceinline__ void copy16(const h16* g, h16* l) {
#if ASYNC_LDS
  h16* gnc = const_cast<h16*>(g);
  __builtin_amdgcn_global_load_async_to_lds_b128(
      (__attribute__((address_space(1))) i4v*)gnc,
      (__attribute__((address_space(3))) i4v*)l, 0, 0);
#else
  *(vu4*)l = *(const vu4*)g;
#endif
}

__device__ __forceinline__ void wait_async_copies() {
#if ASYNC_LDS
#if __has_builtin(__builtin_amdgcn_s_wait_asynccnt)
  __builtin_amdgcn_s_wait_asynccnt(0);
#else
  asm volatile("s_wait_asynccnt 0x0" ::: "memory");
#endif
#endif
}

__device__ __forceinline__ v8f wmma16(v16h a, v16h b, v8f c) {
  // (neg_a, A, neg_b, B, c_mod, C, reuse_a, reuse_b)
  return __builtin_amdgcn_wmma_f32_16x16x32_f16(false, a, false, b,
                                                (short)0, c, false, false);
}

// 16x32 f16 A-fragment (or symmetric B-fragment from N-major LDS).
// ISA 7.12.2: lane l holds row M=l&15; VGPR j holds K pair
//   K = (j&3)*2 + (l>>4)*8 + (j>>2)*16   (pairs contiguous -> b32/b128 reads)
__device__ __forceinline__ v16h load_frag(const h16* lds, int row0, int col0,
                                          int stride) {
  const int lane = threadIdx.x & 31;
  const int r = lane & 15;
  const int hi = lane >> 4;
  const h16* base = lds + (row0 + r) * stride + col0 + hi * 8;
  v16h a;
#pragma unroll
  for (int j = 0; j < 8; ++j) {
    const int kk = (j & 3) * 2 + (j >> 2) * 16;
    h2v p = *(const h2v*)(base + kk);
    a[2 * j]     = p.x;
    a[2 * j + 1] = p.y;
  }
  return a;
}

// stage 64x64 f16 x-tile and centers-tile: 4 async 16B ops per thread
__device__ __forceinline__ void stage_xc(const h16* xh, const h16* ch,
                                         h16* xbuf, h16* cbuf, int m0, int k0,
                                         int d0, int t) {
#pragma unroll
  for (int i = 0; i < 2; ++i) {
    const int idx = t + 256 * i;
    const int r = idx >> 3, seg = (idx & 7) * 8;   // 8 halfs per lane-op
    copy16(xh + (size_t)(m0 + r) * DIM_IN + d0 + seg, xbuf + r * LSTR + seg);
    copy16(ch + (size_t)(k0 + r) * DIM_IN + d0 + seg, cbuf + r * LSTR + seg);
  }
}

// stage 256x64 f16 values chunk (N-major): 8 async 16B ops per thread
__device__ __forceinline__ void stage_v(const h16* vTh, h16* vbuf, int ncol0,
                                        int k0, int t) {
#pragma unroll
  for (int i = 0; i < 8; ++i) {
    const int idx = t + 256 * i;
    const int n = idx >> 3, seg = (idx & 7) * 8;
    copy16(vTh + (size_t)(ncol0 + n) * NK + k0 + seg, vbuf + n * VSTR + seg);
  }
}

// ---- row sum-of-squares over first DIM_IN columns: one wave32 per row -----
__global__ void rbf_rowsumsq(const float* __restrict__ src,
                             float* __restrict__ dst, int rowStride) {
  const int wave = threadIdx.x >> 5;
  const int lane = threadIdx.x & 31;
  const int row  = blockIdx.x * 8 + wave;
  const float* p = src + (size_t)row * rowStride;
  float s = 0.f;
#pragma unroll
  for (int j = 0; j < 8; ++j) {
    vf4 v = *(const vf4*)(p + lane * 4 + j * 128);
    s += v.x * v.x + v.y * v.y + v.z * v.z + v.w * v.w;
  }
#pragma unroll
  for (int off = 16; off > 0; off >>= 1) s += __shfl_down(s, off, 32);
  if (lane == 0) dst[row] = s;
}

// ---- one-time f32 -> f16 row conversion (first DIM_IN cols of each row) ---
__global__ void rbf_cvt_f16(const float* __restrict__ src,
                            h16* __restrict__ dst, int srcStride) {
  const int row = blockIdx.x;
  const int c = threadIdx.x * 4;
  vf4 v = *(const vf4*)(src + (size_t)row * srcStride + c);
  h16* d = dst + (size_t)row * DIM_IN + c;
  d[0] = (h16)v.x; d[1] = (h16)v.y; d[2] = (h16)v.z; d[3] = (h16)v.w;
}

// ---- one-time values f32[k][n] -> f16 vT[n][k] (tiled LDS transpose) ------
__global__ void rbf_cvt_vT(const float* __restrict__ w, h16* __restrict__ vT) {
  __shared__ float tile[64][65];
  const int kb = blockIdx.x * 64;
  const int nb = blockIdx.y * 64;
  const int t = threadIdx.x;
  {
    const int r = t >> 2, c0 = (t & 3) * 16;
    const float* src = w + (size_t)(kb + r) * WROW + DIM_IN + nb + c0;
#pragma unroll
    for (int i = 0; i < 4; ++i) {
      vf4 v = *(const vf4*)(src + 4 * i);
      tile[r][c0 + 4 * i + 0] = v.x;
      tile[r][c0 + 4 * i + 1] = v.y;
      tile[r][c0 + 4 * i + 2] = v.z;
      tile[r][c0 + 4 * i + 3] = v.w;
    }
  }
  __syncthreads();
  {
    const int n = t >> 2, k0c = (t & 3) * 16;
    h16* d = vT + (size_t)(nb + n) * NK + kb + k0c;
#pragma unroll
    for (int i = 0; i < 16; ++i) d[i] = (h16)tile[k0c + i][n];
  }
}

// ---- fused scores+GEMM2 kernel --------------------------------------------
__global__ __launch_bounds__(256)
void rbf_fused(const h16* __restrict__ xh, const h16* __restrict__ ch,
               const h16* __restrict__ vTh, const float* __restrict__ xsq,
               const float* __restrict__ csq, float* __restrict__ out) {
  __shared__ alignas(16) h16 xA[2][BM * LSTR];      // 18432 B (double-buffered)
  __shared__ alignas(16) h16 cB[2][BK * LSTR];      // 18432 B (double-buffered)
  __shared__ alignas(16) h16 Pt[BM * LSTR];         //  9216 B
  __shared__ alignas(16) h16 vB[2][NCHUNK * VSTR];  // 73728 B (double-buffered)
  __shared__ float xs[BM];                          //   256 B
  __shared__ float cs[BK];                          //   256 B  (total ~117.5KB)

  const int m0 = blockIdx.x * BM;
  const int n0 = blockIdx.y * BN;
  const int t    = threadIdx.x;
  const int wave = t >> 5;
  const int lane = t & 31;
  const int r15  = lane & 15;
  const int hi   = lane >> 4;
  // GEMM1 wave mapping: 4 waves over M (16 rows), 2 over N (32 cols)
  const int sm = wave & 3;
  const int sn = wave >> 2;
  // GEMM2 wave mapping: 2 waves over M (32 rows), 4 over N (64 cols/chunk)
  const int wm = wave & 1;
  const int wn = wave >> 1;

  const v8f vzero = {0.f, 0.f, 0.f, 0.f, 0.f, 0.f, 0.f, 0.f};
  v8f acc[16];                                      // [chunk 2][mt 2][nt 4]
#pragma unroll
  for (int i = 0; i < 16; ++i) acc[i] = vzero;

  if (t < BM) xs[t] = xsq[m0 + t];

  // pipeline prologue: first x/centers batch in flight before the main loop
  stage_xc(xh, ch, xA[0], cB[0], m0, 0, 0, t);

  for (int k0 = 0; k0 < NK; k0 += BK) {
    if (t < BK) cs[t] = csq[k0 + t];

    // ---- GEMM1: S(64x64) = x_tile @ centers_tile^T over DIM_IN ----
    v8f sacc0 = vzero, sacc1 = vzero;
#pragma unroll 1
    for (int s = 0; s < NSTEPS; ++s) {
      const int cur = s & 1;
      wait_async_copies();             // batch s landed (per-wave)
      __syncthreads();                 // ... in every wave
      if (s + 1 < NSTEPS) {
        // issue batch s+1 into the other buffer: DMA overlaps WMMA below
        stage_xc(xh, ch, xA[1 - cur], cB[1 - cur], m0, k0, (s + 1) * DSTEP, t);
      } else {
        // last step: start values chunk 0 DMA (overlaps WMMA + exp below)
        stage_v(vTh, vB[0], n0, k0, t);
      }
#pragma unroll
      for (int kk = 0; kk < 2; ++kk) {
        v16h a  = load_frag(xA[cur], 16 * sm,      kk * 32, LSTR);
        v16h b0 = load_frag(cB[cur], 32 * sn,      kk * 32, LSTR);
        v16h b1 = load_frag(cB[cur], 32 * sn + 16, kk * 32, LSTR);
        sacc0 = wmma16(a, b0, sacc0);
        sacc1 = wmma16(a, b1, sacc1);
      }
    }

    // ---- S -> P = exp(-(xsq + csq - 2S)), f16 into LDS (SIGMA = 1) ----
    // (runs while vB chunk-0 DMA is in flight)
    {
      const int m  = 16 * sm + 8 * hi;
      const int nc = 32 * sn + r15;
#pragma unroll
      for (int r = 0; r < 8; ++r) {
        const float xv = xs[m + r];
        const float d0v = xv + cs[nc]      - 2.f * sacc0[r];
        const float d1v = xv + cs[nc + 16] - 2.f * sacc1[r];
        Pt[(m + r) * LSTR + nc]      = (h16)__expf(-d0v);
        Pt[(m + r) * LSTR + nc + 16] = (h16)__expf(-d1v);
      }
    }

    // ---- GEMM2: acc += P(64x64) @ values(64x512), 2 pipelined chunks ----
#pragma unroll
    for (int c = 0; c < 2; ++c) {
      wait_async_copies();             // vB chunk c landed
      __syncthreads();                 // P visible + all waves' DMA done
      if (c == 0) {
        stage_v(vTh, vB[1], n0 + NCHUNK, k0, t);         // chunk 1 DMA
      } else if (k0 + BK < NK) {
        stage_xc(xh, ch, xA[0], cB[0], m0, k0 + BK, 0, t); // next k-iter DMA
      }
#pragma unroll
      for (int kk = 0; kk < 2; ++kk) {
        v16h a0 = load_frag(Pt, 32 * wm,      kk * 32, LSTR);
        v16h a1 = load_frag(Pt, 32 * wm + 16, kk * 32, LSTR);
#pragma unroll
        for (int nt = 0; nt < 4; ++nt) {
          v16h b = load_frag(vB[c], 64 * wn + 16 * nt, kk * 32, VSTR);
          acc[(c * 2 + 0) * 4 + nt] = wmma16(a0, b, acc[(c * 2 + 0) * 4 + nt]);
          acc[(c * 2 + 1) * 4 + nt] = wmma16(a1, b, acc[(c * 2 + 1) * 4 + nt]);
        }
      }
    }
  }

  // ---- epilogue: C-layout (ISA 7.12.2): VGPR r -> M = r + 8*(lane>>4) ----
#pragma unroll
  for (int c = 0; c < 2; ++c)
#pragma unroll
    for (int mt = 0; mt < 2; ++mt)
#pragma unroll
      for (int nt = 0; nt < 4; ++nt) {
        v8f a = acc[(c * 2 + mt) * 4 + nt];
        const int rowb = m0 + 32 * wm + 16 * mt + 8 * hi;
        const int col  = n0 + c * NCHUNK + 64 * wn + 16 * nt + r15;
#pragma unroll
        for (int r = 0; r < 8; ++r)
          out[(size_t)(rowb + r) * DIM_OUT + col] = a[r];
      }
}

extern "C" void kernel_launch(void* const* d_in, const int* in_sizes, int n_in,
                              void* d_out, int out_size, void* d_ws,
                              size_t ws_size, hipStream_t stream) {
  const float* x = (const float*)d_in[0];       // (8192, 1024) f32
  const float* w = (const float*)d_in[1];       // (8192, 2048) f32
  float* out = (float*)d_out;                   // (8192, 1024) f32

  // workspace layout: 64KB f32 norms + 48MB f16 converted operands
  float* xsq = (float*)d_ws;                    // 8192 f32
  float* csq = xsq + NB;                        // 8192 f32
  h16* xh  = (h16*)(csq + NK);                  // 8192x1024 f16 (16MB)
  h16* ch  = xh + (size_t)NB * DIM_IN;          // 8192x1024 f16 (16MB)
  h16* vTh = ch + (size_t)NK * DIM_IN;          // 1024x8192 f16 (16MB), [n][k]

  rbf_rowsumsq<<<NB / 8, 256, 0, stream>>>(x, xsq, DIM_IN);   // |x_i|^2
  rbf_rowsumsq<<<NK / 8, 256, 0, stream>>>(w, csq, WROW);     // |c_k|^2
  rbf_cvt_f16<<<NB, 256, 0, stream>>>(x, xh, DIM_IN);
  rbf_cvt_f16<<<NK, 256, 0, stream>>>(w, ch, WROW);
  dim3 tgrid(NK / 64, DIM_OUT / 64);
  rbf_cvt_vT<<<tgrid, 256, 0, stream>>>(w, vTh);

  dim3 grid(NB / BM, DIM_OUT / BN);             // 128 x 2 workgroups
  rbf_fused<<<grid, 256, 0, stream>>>(xh, ch, vTh, xsq, csq, out);
}
